// SimpleLSTM_31525059953341
// MI455X (gfx1250) — compile-verified, matrix operand order checked by reading, not run on hardware
//
#include <hip/hip_runtime.h>

// ---------------------------------------------------------------------------
// SimpleLSTM for MI455X (gfx1250, wave32, WMMA).
// gates = x_t @ W_ih^T + h @ W_hh^T + (b_ih + b_hh), fused with the
// sigmoid/tanh/cell update, using v_wmma_f32_16x16x32_bf16 (fp32 accumulate).
// K-loop is software-pipelined (register double-buffered fragments) so the
// global_load_b128s for step k+1 overlap the WMMAs of step k.
// ---------------------------------------------------------------------------

typedef __bf16 bf16_t;
typedef __attribute__((ext_vector_type(8)))  __bf16 v8bf;
typedef __attribute__((ext_vector_type(16))) __bf16 v16bf;
typedef __attribute__((ext_vector_type(8)))  float  v8f;

#define LB 128   // batch
#define LS 512   // seq len
#define LI 512   // input size
#define LH 1024  // hidden size
#define LG 4096  // 4*H

// ---- helpers --------------------------------------------------------------

__device__ __forceinline__ bf16_t f2bf(float f) {
    unsigned u = __float_as_uint(f);
    unsigned r = (u + 0x7FFFu + ((u >> 16) & 1u)) >> 16;  // round-nearest-even
    unsigned short s = (unsigned short)r;
    return *reinterpret_cast<bf16_t*>(&s);
}

__device__ __forceinline__ float sigmoidf_fast(float x) {
    return 1.0f / (1.0f + __expf(-x));
}
__device__ __forceinline__ float tanhf_fast(float x) {
    return 2.0f / (1.0f + __expf(-2.0f * x)) - 1.0f;
}

// Load one 16x32 bf16 fragment (row-major over K).
// p0 points at element (row 0, k of this K-chunk); rows `stride` elems apart.
// Lane l (0..15): row l, K = {0..7, 16..23}; lane l+16: row l, K = {8..15, 24..31}
// (CDNA5 16-bit A-matrix VGPR layout; B uses the symmetric column layout,
// which for gates = act @ W^T means B-lanes read W's rows contiguously).
__device__ __forceinline__ v16bf load_frag(const bf16_t* __restrict__ p0,
                                           size_t stride, int lane) {
    const bf16_t* p = p0 + (size_t)(lane & 15) * stride + ((lane >> 4) << 3);
    v8bf lo = *reinterpret_cast<const v8bf*>(p);
    v8bf hi = *reinterpret_cast<const v8bf*>(p + 16);
    return __builtin_shufflevector(lo, hi, 0, 1, 2, 3, 4, 5, 6, 7,
                                           8, 9, 10, 11, 12, 13, 14, 15);
}

__device__ __forceinline__ v8f wmma_bf16(v16bf a, v16bf b, v8f c) {
    return __builtin_amdgcn_wmma_f32_16x16x32_bf16(
        /*neg_a=*/false, a, /*neg_b=*/false, b,
        /*c_mod=*/(short)0, c, /*reuse_a=*/false, /*reuse_b=*/false);
}

// Pipelined GEMM segment: acc += A(16xK) * B(Kx16) for 4 gate B-panels.
// Loads for iteration k+1 are issued before the WMMAs of iteration k, so the
// backend can wait on a partial loadcnt and keep VMEM in flight under WMMA.
template <int KLEN>
__device__ __forceinline__ void gemm_seg(const bf16_t* __restrict__ abase, size_t astride,
                                         const bf16_t* __restrict__ b0,
                                         const bf16_t* __restrict__ b1,
                                         const bf16_t* __restrict__ b2,
                                         const bf16_t* __restrict__ b3,
                                         size_t bstride, int lane,
                                         v8f& ai, v8f& af, v8f& ag, v8f& ao) {
    v16bf a_c  = load_frag(abase, astride, lane);
    v16bf bi_c = load_frag(b0, bstride, lane);
    v16bf bf_c = load_frag(b1, bstride, lane);
    v16bf bg_c = load_frag(b2, bstride, lane);
    v16bf bo_c = load_frag(b3, bstride, lane);
#pragma unroll 2
    for (int k = 32; k < KLEN; k += 32) {
        v16bf a_n  = load_frag(abase + k, astride, lane);
        v16bf bi_n = load_frag(b0 + k, bstride, lane);
        v16bf bf_n = load_frag(b1 + k, bstride, lane);
        v16bf bg_n = load_frag(b2 + k, bstride, lane);
        v16bf bo_n = load_frag(b3 + k, bstride, lane);
        ai = wmma_bf16(a_c, bi_c, ai);
        af = wmma_bf16(a_c, bf_c, af);
        ag = wmma_bf16(a_c, bg_c, ag);
        ao = wmma_bf16(a_c, bo_c, ao);
        a_c = a_n; bi_c = bi_n; bf_c = bf_n; bg_c = bg_n; bo_c = bo_n;
    }
    ai = wmma_bf16(a_c, bi_c, ai);
    af = wmma_bf16(a_c, bf_c, af);
    ag = wmma_bf16(a_c, bg_c, ag);
    ao = wmma_bf16(a_c, bo_c, ao);
}

// ---- init: fp32->bf16 conversions, bias fold, state zeroing ---------------

__global__ void lstm_init_kernel(const float* __restrict__ x,
                                 const float* __restrict__ wih,
                                 const float* __restrict__ whh,
                                 const float* __restrict__ bih,
                                 const float* __restrict__ bhh,
                                 bf16_t* __restrict__ x_bf,
                                 bf16_t* __restrict__ wih_bf,
                                 bf16_t* __restrict__ whh_bf,
                                 float*  __restrict__ bias,
                                 bf16_t* __restrict__ h0,
                                 bf16_t* __restrict__ h1,
                                 float*  __restrict__ c_state,
                                 float*  __restrict__ h_final) {
    const size_t tid    = (size_t)blockIdx.x * blockDim.x + threadIdx.x;
    const size_t stride = (size_t)gridDim.x * blockDim.x;

    for (size_t k = tid; k < (size_t)LB * LS * LI; k += stride) x_bf[k]   = f2bf(x[k]);
    for (size_t k = tid; k < (size_t)LG * LI;      k += stride) wih_bf[k] = f2bf(wih[k]);
    for (size_t k = tid; k < (size_t)LG * LH;      k += stride) whh_bf[k] = f2bf(whh[k]);
    for (size_t k = tid; k < (size_t)LG;           k += stride) bias[k]   = bih[k] + bhh[k];
    for (size_t k = tid; k < (size_t)LB * LH;      k += stride) {
        h0[k] = f2bf(0.0f);
        h1[k] = f2bf(0.0f);
        c_state[k] = 0.0f;
        h_final[k] = 0.0f;
    }
}

// ---- one recurrence step ---------------------------------------------------
// Grid: 64 blocks (4 batch-tiles x 16 hidden-tiles), 256 threads = 8 waves.
// Each wave owns one 16x16 (batch x hidden) tile and accumulates i/f/g/o
// gate tiles in matching lanes -> in-register elementwise LSTM update.

__global__ __launch_bounds__(256, 1) void lstm_step_kernel(
    const bf16_t* __restrict__ wih_bf,   // [4H, I]
    const bf16_t* __restrict__ whh_bf,   // [4H, H]
    const float*  __restrict__ bias,     // [4H]
    const bf16_t* __restrict__ x_bf,     // [B, S, I]
    const bf16_t* __restrict__ h_in,     // [B, H]
    bf16_t*       __restrict__ h_out,    // [B, H]
    float*        __restrict__ outputs,  // [B, S, H]
    float*        __restrict__ h_final,  // [B, H]
    float*        __restrict__ c_state,  // [B, H]
    int t) {
    const int lane = threadIdx.x & 31;
    const int wave = threadIdx.x >> 5;
    const int bm   = blockIdx.x >> 4;   // 0..3  -> batch tile of 32
    const int bn   = blockIdx.x & 15;   // 0..15 -> hidden tile of 64
    const int msub = wave >> 2;         // 0..1
    const int nsub = wave & 3;          // 0..3
    const int m0   = bm * 32 + msub * 16;  // batch row base
    const int j0   = bn * 64 + nsub * 16;  // hidden col base

    v8f acc_i = {}, acc_f = {}, acc_g = {}, acc_o = {};

    // input projection part: K over I, A = x[:, t, :]
    gemm_seg<LI>(x_bf + ((size_t)m0 * LS + t) * LI, (size_t)LS * LI,
                 wih_bf + ((size_t)0 * LH + j0) * LI,
                 wih_bf + ((size_t)1 * LH + j0) * LI,
                 wih_bf + ((size_t)2 * LH + j0) * LI,
                 wih_bf + ((size_t)3 * LH + j0) * LI,
                 (size_t)LI, lane, acc_i, acc_f, acc_g, acc_o);

    // hidden projection part: K over H, A = h_in (W_hh is L2-resident)
    gemm_seg<LH>(h_in + (size_t)m0 * LH, (size_t)LH,
                 whh_bf + ((size_t)0 * LH + j0) * LH,
                 whh_bf + ((size_t)1 * LH + j0) * LH,
                 whh_bf + ((size_t)2 * LH + j0) * LH,
                 whh_bf + ((size_t)3 * LH + j0) * LH,
                 (size_t)LH, lane, acc_i, acc_f, acc_g, acc_o);

    // --- fused gate nonlinearities + cell/hidden update (in-register) ---
    // C/D layout: VGPR r, lanes 0-15 -> (M=r, N=lane); lanes 16-31 -> (M=r+8).
    const int jcol = j0 + (lane & 15);
    const int mofs = (lane >> 4) * 8;
    const float b_i = bias[0 * LH + jcol];
    const float b_f = bias[1 * LH + jcol];
    const float b_g = bias[2 * LH + jcol];
    const float b_o = bias[3 * LH + jcol];

#pragma unroll
    for (int r = 0; r < 8; ++r) {
        const int b = m0 + mofs + r;  // batch index
        const float gi = sigmoidf_fast(acc_i[r] + b_i);
        const float gf = sigmoidf_fast(acc_f[r] + b_f);
        const float gg = tanhf_fast(acc_g[r] + b_g);
        const float go = sigmoidf_fast(acc_o[r] + b_o);

        const size_t sidx = (size_t)b * LH + jcol;
        const float c_old = c_state[sidx];
        const float c_new = gf * c_old + gi * gg;
        const float h_new = go * tanhf_fast(c_new);

        c_state[sidx] = c_new;
        h_final[sidx] = h_new;
        h_out[sidx]   = f2bf(h_new);
        outputs[((size_t)b * LS + t) * LH + jcol] = h_new;
    }
}

// ---- host launch -----------------------------------------------------------

extern "C" void kernel_launch(void* const* d_in, const int* in_sizes, int n_in,
                              void* d_out, int out_size, void* d_ws, size_t ws_size,
                              hipStream_t stream) {
    const float* x   = (const float*)d_in[0];  // [B,S,I]
    const float* wih = (const float*)d_in[1];  // [4H,I]
    const float* whh = (const float*)d_in[2];  // [4H,H]
    const float* bih = (const float*)d_in[3];  // [4H]
    const float* bhh = (const float*)d_in[4];  // [4H]

    // d_out = outputs [B,S,H] ++ h_t [B,H] ++ c_t [B,H]  (fp32)
    float* outputs = (float*)d_out;
    float* h_final = outputs + (size_t)LB * LS * LH;
    float* c_state = h_final + (size_t)LB * LH;

    // workspace carve-up (all offsets well aligned)
    char* ws = (char*)d_ws;
    bf16_t* wih_bf = (bf16_t*)ws;                       // 4 MiB
    ws += (size_t)LG * LI * sizeof(bf16_t);
    bf16_t* whh_bf = (bf16_t*)ws;                       // 8 MiB
    ws += (size_t)LG * LH * sizeof(bf16_t);
    bf16_t* x_bf = (bf16_t*)ws;                         // 64 MiB
    ws += (size_t)LB * LS * LI * sizeof(bf16_t);
    float* bias = (float*)ws;                           // 16 KiB
    ws += (size_t)LG * sizeof(float);
    bf16_t* hbuf0 = (bf16_t*)ws;                        // 256 KiB
    ws += (size_t)LB * LH * sizeof(bf16_t);
    bf16_t* hbuf1 = (bf16_t*)ws;                        // 256 KiB

    lstm_init_kernel<<<2048, 256, 0, stream>>>(x, wih, whh, bih, bhh,
                                               x_bf, wih_bf, whh_bf, bias,
                                               hbuf0, hbuf1, c_state, h_final);

    bf16_t* hb[2] = {hbuf0, hbuf1};
    for (int t = 0; t < LS; ++t) {
        lstm_step_kernel<<<64, 256, 0, stream>>>(
            wih_bf, whh_bf, bias, x_bf,
            hb[t & 1], hb[(t + 1) & 1],
            outputs, h_final, c_state, t);
    }
}
